// Conv_Attention_11879879544366
// MI455X (gfx1250) — matrix operand order repaired and checked by reading, not compiled, not run on hardware
//
#include <hip/hip_runtime.h>
#include <hip/hip_bf16.h>

// ---------------------------------------------------------------------------
// CDNA5 (gfx1250) windowed 3D attention, bf16 WMMA pipeline.
// ---------------------------------------------------------------------------

typedef __bf16 bf16_t;
typedef __attribute__((ext_vector_type(16))) __bf16 v16bf;
typedef __attribute__((ext_vector_type(8)))  __bf16 v8bf;
typedef __attribute__((ext_vector_type(8)))  float  v8f;

#define NWIN   288      // b*gx*gy = 2*12*12
#define NTOK   294      // 6*7*7
#define NPAD   304      // 19 * 16
#define NTT    19       // token tiles of 16
#define DIM    256
#define HEADS  8
#define DH     32
#define QSCALE 0.17677669529663687f   // 32^-0.5

__device__ __forceinline__ bf16_t to_bf16(float f) {
  unsigned u = __builtin_bit_cast(unsigned, f);
  u += 0x7FFFu + ((u >> 16) & 1u);              // round-to-nearest-even
  unsigned short h = (unsigned short)(u >> 16);
  return __builtin_bit_cast(bf16_t, h);
}

// A-matrix fragment (16x32, 16-bit): lane half h owns K=16h..16h+15,
// contiguous.  base points at [row][16*half].
__device__ __forceinline__ v16bf load_fragA(const bf16_t* base) {
  v8bf lo = *(const v8bf*)(base);
  v8bf hi = *(const v8bf*)(base + 8);
  v16bf r;
#pragma unroll
  for (int e = 0; e < 8; ++e) { r[e] = lo[e]; r[8 + e] = hi[e]; }
  return r;
}

// B-matrix fragment (32x16, 16-bit): lanes 0-15 hold K=0..15, lanes 16-31 hold
// K=16..31, contiguous.  base points at [col][16*half].
__device__ __forceinline__ v16bf load_fragB(const bf16_t* base) {
  v8bf lo = *(const v8bf*)(base);
  v8bf hi = *(const v8bf*)(base + 8);
  v16bf r;
#pragma unroll
  for (int e = 0; e < 8; ++e) { r[e] = lo[e]; r[8 + e] = hi[e]; }
  return r;
}

#define WMMA_BF16(a, b, c) \
  __builtin_amdgcn_wmma_f32_16x16x32_bf16(false, (a), false, (b), (short)0, (c), false, false)

// ---------------------------------------------------------------------------
// Kernel A: weight transposes -> bf16, column-major-K layout.
// grid.x = 1024 (768 for Wq|Wkv columns, 256 for Wout), block = 256 (= k).
// ---------------------------------------------------------------------------
__global__ __launch_bounds__(256) void prep_weights_kernel(
    const float* __restrict__ Wq, const float* __restrict__ Wkv,
    const float* __restrict__ Wout, bf16_t* __restrict__ Wt,
    bf16_t* __restrict__ Wt2) {
  const int n = blockIdx.x;
  const int k = threadIdx.x;
  if (n < 256)      Wt[n * 256 + k]          = to_bf16(Wq[k * 256 + n]);
  else if (n < 768) Wt[n * 256 + k]          = to_bf16(Wkv[k * 512 + (n - 256)]);
  else              Wt2[(n - 768) * 256 + k] = to_bf16(Wout[k * 256 + (n - 768)]);
}

// ---------------------------------------------------------------------------
// Kernel B: expand relative-position bias to [8][304][304] f32 (padded zeros).
// grid.x = 8*304 (head,i), block = 128, j strided.
// ---------------------------------------------------------------------------
__global__ __launch_bounds__(128) void prep_bias_kernel(
    const float* __restrict__ bias_table, float* __restrict__ bias_full) {
  const int hi = blockIdx.x;
  const int h = hi / NPAD, i = hi % NPAD;
  const int li = i / 49, rmi = i % 49, wi1 = rmi / 7, wi2 = rmi % 7;
  for (int j = threadIdx.x; j < NPAD; j += blockDim.x) {
    float v = 0.f;
    if (i < NTOK && j < NTOK) {
      const int lj = j / 49, rmj = j % 49, wj1 = rmj / 7, wj2 = rmj % 7;
      const int idx = (li - lj + 5) * 169 + (wi1 - wj1 + 6) * 13 + (wi2 - wj2 + 6);
      v = bias_table[idx * HEADS + h];
    }
    bias_full[((size_t)h * NPAD + i) * NPAD + j] = v;
  }
}

// ---------------------------------------------------------------------------
// Kernel C: fused QKV projection GEMM (84672 x 256) x (256 x 768) in bf16 WMMA.
// grid = (288*19, 6), block = 256 (8 waves, each a 16x16 tile; 128 cols/block).
// Writes q_ws/k_ws: [w][h][tpad][dh] bf16 (q pre-scaled), v_ws: [w][h][dh][tpad].
// ---------------------------------------------------------------------------
__global__ __launch_bounds__(256) void qkv_proj_kernel(
    const float* __restrict__ x, const float* __restrict__ bq,
    const float* __restrict__ bkv, const bf16_t* __restrict__ Wt,
    bf16_t* __restrict__ q_ws, bf16_t* __restrict__ k_ws,
    bf16_t* __restrict__ v_ws) {
  __shared__ __align__(16) bf16_t As[16][256];

  const int wtile = blockIdx.x;
  const int w = wtile / NTT, tt = wtile % NTT;
  const int bb = w / 144, gxy = w % 144, gx = gxy / 12, gy = gxy % 12;

  {  // cooperative A-tile load: 16 rows x 256 K, fp32 -> bf16 -> LDS
    const int tid = threadIdx.x;
    const int row = tid >> 4;
    const int c0 = (tid & 15) * 16;
    const int t = tt * 16 + row;
    if (t < NTOK) {
      const int l = t / 49, rm = t % 49, w1 = rm / 7, w2 = rm % 7;
      const size_t ga =
          ((((((size_t)bb * 6 + l) * 12 + gx) * 12 + gy) * 7 + w1) * 7 + w2) * DIM;
      const float4* src = (const float4*)(x + ga + c0);
#pragma unroll
      for (int v = 0; v < 4; ++v) {
        float4 f = src[v];
        As[row][c0 + v * 4 + 0] = to_bf16(f.x);
        As[row][c0 + v * 4 + 1] = to_bf16(f.y);
        As[row][c0 + v * 4 + 2] = to_bf16(f.z);
        As[row][c0 + v * 4 + 3] = to_bf16(f.w);
      }
    } else {
      const bf16_t z = to_bf16(0.f);
#pragma unroll
      for (int v = 0; v < 16; ++v) As[row][c0 + v] = z;
    }
  }
  __syncthreads();

  const int wave  = threadIdx.x >> 5;
  const int lane  = threadIdx.x & 31;
  const int lhalf = lane >> 4;
  const int l15   = lane & 15;
  const int nbase = blockIdx.y * 128 + wave * 16;

  v8f acc = {};
#pragma unroll
  for (int kk = 0; kk < 8; ++kk) {
    const int k0 = kk * 32;
    v16bf a = load_fragA(&As[l15][k0 + lhalf * 16]);
    v16bf b = load_fragB(Wt + (size_t)(nbase + l15) * 256 + k0 + lhalf * 16);
    acc = WMMA_BF16(a, b, acc);
  }

  const int nc = nbase + l15;  // output column owned by this lane
  const float biasv = (nc < 256) ? bq[nc] : bkv[nc - 256];
#pragma unroll
  for (int r = 0; r < 8; ++r) {
    const int t = tt * 16 + r + lhalf * 8;
    const float val = acc[r] + biasv;
    if (nc < 256) {
      const int h = nc >> 5, c = nc & 31;
      q_ws[(((size_t)w * HEADS + h) * NPAD + t) * DH + c] = to_bf16(val * QSCALE);
    } else if (nc < 512) {
      const int h = (nc - 256) >> 5, c = (nc - 256) & 31;
      k_ws[(((size_t)w * HEADS + h) * NPAD + t) * DH + c] = to_bf16(val);
    } else {
      const int h = (nc - 512) >> 5, c = (nc - 512) & 31;
      v_ws[(((size_t)w * HEADS + h) * DH + c) * NPAD + t] = to_bf16(val);
    }
  }
}

// ---------------------------------------------------------------------------
// Kernel D: flash attention per (window, head, q-tile).  grid = 288*8*19,
// block = 32 (one wave).  10 j-steps of 32: 2 WMMA sim + 2 WMMA p@V each,
// with global_prefetch of the next step's K/V fragments overlapping softmax.
// ---------------------------------------------------------------------------
__global__ __launch_bounds__(32) void attn_kernel(
    const bf16_t* __restrict__ q_ws, const bf16_t* __restrict__ k_ws,
    const bf16_t* __restrict__ v_ws, const float* __restrict__ bias_full,
    bf16_t* __restrict__ o_ws) {
  __shared__ __align__(16) bf16_t pbuf[16][32];

  const int blk = blockIdx.x;
  const int qt = blk % NTT;
  const int wh = blk / NTT;
  const int h = wh & 7;
  const int w = wh >> 3;

  const int lane  = threadIdx.x & 31;
  const int lhalf = lane >> 4;
  const int l15   = lane & 15;

  const bf16_t* qb = q_ws + ((size_t)w * HEADS + h) * NPAD * DH;
  const bf16_t* kb = k_ws + ((size_t)w * HEADS + h) * NPAD * DH;
  const bf16_t* vb = v_ws + ((size_t)w * HEADS + h) * DH * NPAD;
  const float*  bf = bias_full + (size_t)h * NPAD * NPAD;

  const v16bf qa = load_fragA(qb + (size_t)(qt * 16 + l15) * DH + lhalf * 16);

  v8f o0 = {}, o1 = {};
  float m_[8], l_[8];
#pragma unroll
  for (int r = 0; r < 8; ++r) { m_[r] = -1e30f; l_[r] = 0.f; }

  for (int jt = 0; jt < 10; ++jt) {
    const int j0 = jt * 32;
    // sim tiles: cols j0+l15 and j0+16+l15  (padded reads stay inside d_ws)
    v16bf kf0 = load_fragB(kb + (size_t)(j0 + l15) * DH + lhalf * 16);
    v16bf kf1 = load_fragB(kb + (size_t)(j0 + 16 + l15) * DH + lhalf * 16);
    v8f s0 = {}, s1 = {};
    s0 = WMMA_BF16(qa, kf0, s0);
    s1 = WMMA_BF16(qa, kf1, s1);

    // Prefetch next j-step's K/V fragments (global_prefetch_b8) so the L2
    // round-trip overlaps the softmax VALU chain below.
    if (jt + 1 < 10) {
      const int jn = (jt + 1) * 32;
      __builtin_prefetch(kb + (size_t)(jn + l15) * DH + lhalf * 16, 0, 3);
      __builtin_prefetch(kb + (size_t)(jn + 16 + l15) * DH + lhalf * 16, 0, 3);
      __builtin_prefetch(vb + (size_t)l15 * NPAD + jn + lhalf * 16, 0, 3);
      __builtin_prefetch(vb + (size_t)(16 + l15) * NPAD + jn + lhalf * 16, 0, 3);
    }

    const int jA = j0 + l15, jB = j0 + 16 + l15;
    float sv0[8], sv1[8];
#pragma unroll
    for (int r = 0; r < 8; ++r) {
      const int i = qt * 16 + r + lhalf * 8;
      sv0[r] = (jA < NTOK) ? s0[r] + bf[(size_t)i * NPAD + jA] : -1e30f;
      sv1[r] = (jB < NTOK) ? s1[r] + bf[(size_t)i * NPAD + jB] : -1e30f;
    }
    // online softmax per row (rows live in 16-lane halves of the wave)
#pragma unroll
    for (int r = 0; r < 8; ++r) {
      float mx = fmaxf(sv0[r], sv1[r]);
#pragma unroll
      for (int msk = 1; msk < 16; msk <<= 1) mx = fmaxf(mx, __shfl_xor(mx, msk, 32));
      const float mnew = fmaxf(m_[r], mx);
      const float corr = __expf(m_[r] - mnew);
      const float p0 = __expf(sv0[r] - mnew);
      const float p1 = __expf(sv1[r] - mnew);
      float ps = p0 + p1;
#pragma unroll
      for (int msk = 1; msk < 16; msk <<= 1) ps += __shfl_xor(ps, msk, 32);
      l_[r] = l_[r] * corr + ps;
      m_[r] = mnew;
      o0[r] *= corr;
      o1[r] *= corr;
      const int iloc = r + lhalf * 8;
      pbuf[iloc][l15]      = to_bf16(p0);
      pbuf[iloc][16 + l15] = to_bf16(p1);
    }
    // same-wave LDS RAW: DS ops are in-order; stop compiler reordering only.
    __asm__ __volatile__("" ::: "memory");

    const v16bf pa  = load_fragA(&pbuf[l15][lhalf * 16]);
    const v16bf vf0 = load_fragB(vb + (size_t)l15 * NPAD + j0 + lhalf * 16);
    const v16bf vf1 = load_fragB(vb + (size_t)(16 + l15) * NPAD + j0 + lhalf * 16);
    o0 = WMMA_BF16(pa, vf0, o0);
    o1 = WMMA_BF16(pa, vf1, o1);
  }

  const size_t obase = (size_t)w * NPAD * DIM + h * DH;
#pragma unroll
  for (int r = 0; r < 8; ++r) {
    const float inv = 1.f / fmaxf(l_[r], 1e-20f);
    const int t = qt * 16 + r + lhalf * 8;
    o_ws[obase + (size_t)t * DIM + l15]      = to_bf16(o0[r] * inv);
    o_ws[obase + (size_t)t * DIM + 16 + l15] = to_bf16(o1[r] * inv);
  }
}

// ---------------------------------------------------------------------------
// Kernel E: output projection (M x 256) x (256 x 256) + permuted f32 store.
// grid = (288*19, 2), block = 256 (8 waves x 16 cols = 128 cols per block).
// ---------------------------------------------------------------------------
__global__ __launch_bounds__(256) void out_proj_kernel(
    const bf16_t* __restrict__ o_ws, const bf16_t* __restrict__ Wt2,
    float* __restrict__ out) {
  const int wtile = blockIdx.x;
  const int w = wtile / NTT, tt = wtile % NTT;
  const int wave  = threadIdx.x >> 5;
  const int lane  = threadIdx.x & 31;
  const int lhalf = lane >> 4;
  const int l15   = lane & 15;
  const int nbase = blockIdx.y * 128 + wave * 16;

  const bf16_t* arow = o_ws + (size_t)(w * NPAD + tt * 16 + l15) * DIM;
  v8f acc = {};
#pragma unroll
  for (int kk = 0; kk < 8; ++kk) {
    const int k0 = kk * 32;
    v16bf a = load_fragA(arow + k0 + lhalf * 16);
    v16bf b = load_fragB(Wt2 + (size_t)(nbase + l15) * 256 + k0 + lhalf * 16);
    acc = WMMA_BF16(a, b, acc);
  }

  const int bb = w / 144, gxy = w % 144, gx = gxy / 12, gy = gxy % 12;
#pragma unroll
  for (int r = 0; r < 8; ++r) {
    const int t = tt * 16 + r + lhalf * 8;
    if (t < NTOK) {
      const int l = t / 49, rm = t % 49, w1 = rm / 7, w2 = rm % 7;
      const size_t ga =
          ((((((size_t)bb * 6 + l) * 12 + gx) * 12 + gy) * 7 + w1) * 7 + w2) * DIM;
      out[ga + nbase + l15] = acc[r];
    }
  }
}

// ---------------------------------------------------------------------------
extern "C" void kernel_launch(void* const* d_in, const int* in_sizes, int n_in,
                              void* d_out, int out_size, void* d_ws, size_t ws_size,
                              hipStream_t stream) {
  (void)in_sizes; (void)n_in; (void)out_size; (void)ws_size;
  const float* x          = (const float*)d_in[0];
  const float* Wq         = (const float*)d_in[1];
  const float* bq         = (const float*)d_in[2];
  const float* Wkv        = (const float*)d_in[3];
  const float* bkv        = (const float*)d_in[4];
  const float* Wout       = (const float*)d_in[5];
  const float* bias_table = (const float*)d_in[6];
  float* out = (float*)d_out;

  // Workspace carve-up (order matters: padded tail reads from k_ws/v_ws must
  // land in a later buffer, never past the end of d_ws).
  char* p = (char*)d_ws;
  auto carve = [&](size_t bytes) {
    char* r = p;
    p += (bytes + 255) & ~(size_t)255;
    return r;
  };
  bf16_t* Wt        = (bf16_t*)carve((size_t)768 * 256 * sizeof(bf16_t));
  bf16_t* Wt2       = (bf16_t*)carve((size_t)256 * 256 * sizeof(bf16_t));
  float*  bias_full = (float*) carve((size_t)HEADS * NPAD * NPAD * sizeof(float));
  bf16_t* q_ws      = (bf16_t*)carve((size_t)NWIN * HEADS * NPAD * DH * sizeof(bf16_t));
  bf16_t* k_ws      = (bf16_t*)carve((size_t)NWIN * HEADS * NPAD * DH * sizeof(bf16_t));
  bf16_t* v_ws      = (bf16_t*)carve((size_t)NWIN * HEADS * DH * NPAD * sizeof(bf16_t));
  bf16_t* o_ws      = (bf16_t*)carve((size_t)NWIN * NPAD * DIM * sizeof(bf16_t));

  prep_weights_kernel<<<1024, 256, 0, stream>>>(Wq, Wkv, Wout, Wt, Wt2);
  prep_bias_kernel<<<HEADS * NPAD, 128, 0, stream>>>(bias_table, bias_full);
  qkv_proj_kernel<<<dim3(NWIN * NTT, 6), 256, 0, stream>>>(x, bq, bkv, Wt, q_ws,
                                                           k_ws, v_ws);
  attn_kernel<<<NWIN * HEADS * NTT, 32, 0, stream>>>(q_ws, k_ws, v_ws, bias_full,
                                                     o_ws);
  out_proj_kernel<<<dim3(NWIN * NTT, 2), 256, 0, stream>>>(o_ws, Wt2, out);
}
